// ReppointsRetinaNet_12283606467836
// MI455X (gfx1250) — matrix operand order repaired and checked by reading, not compiled, not run on hardware
//
#include <hip/hip_runtime.h>
#include <math.h>

// ---------------------------------------------------------------------------
// CDNA5 / gfx1250 WMMA types and fragment layouts (CDNA5 ISA, wave32):
//   A (16x32 f16): VGPR j holds K = (j>=4?16:0) + 8*half + 2*(j&3), M = lane&15
//   B (32x16 f16): VGPR j holds K = (j>=4?16:0) + 8*half + 2*(j&3),                  N = lane&15
//   C/D (16x16 f32): VGPR v holds M = v + 8*half,                   N = lane&15
// ---------------------------------------------------------------------------
typedef __attribute__((ext_vector_type(16))) _Float16 v16h;
typedef __attribute__((ext_vector_type(8)))  float    v8f;
typedef __attribute__((ext_vector_type(4)))  unsigned int u32x4;
typedef __attribute__((ext_vector_type(8)))  int      i32x8;
typedef __attribute__((ext_vector_type(4)))  int      i32x4;

union FragU { unsigned int u[8]; v16h v; };

__device__ __forceinline__ constexpr int a_koff(int j, int half) {
    return ((j >= 4) ? 16 : 0) + half * 8 + 2 * (j & 3);
}
__device__ __forceinline__ constexpr int b_koff(int j, int half) {
    return ((j >= 4) ? 16 : 0) + half * 8 + 2 * (j & 3);
}

#define LDSP 36    // padded A-tile row stride (halves)
// Weight slab in LDS: 64 rows x 256 halves, padded 8B per 64B chunk by the TDM:
//   element k of row n at halves offset n*288 + (k>>5)*36 + (k&31)
#define WROW 288

#if __has_builtin(__builtin_amdgcn_tensor_load_to_lds)
#define HAVE_TDM 1
#endif

// ---------------------------------------------------------------------------
// Tensor Data Mover: DMA one tap's weight slab [64 n x 256 k] f16 from global
// into LDS, hardware-padded (pad_interval=3 -> every 16 dwords, pad_amount=1
// -> 2 dwords) to the WROW layout. One TDM issue per workgroup (wave 0),
// completion via s_wait_tensorcnt, then workgroup barrier (caller).
// D# layout per CDNA5 ISA.
// ---------------------------------------------------------------------------
__device__ __forceinline__ void stage_w_tap(_Float16* Wsh, const _Float16* gsrc, int tid)
{
#if defined(HAVE_TDM)
    if (__builtin_amdgcn_readfirstlane(tid) == 0) {   // wave 0 only (scalar branch)
        const unsigned long long ga = (unsigned long long)(size_t)gsrc;
        const unsigned lds = (unsigned)(size_t)Wsh;   // LDS offset (low 32 bits)
        u32x4 g0;
        g0.x = 1u;                                    // count=1 (valid user D#)
        g0.y = lds;                                   // lds_addr
        g0.z = (unsigned)(ga & 0xffffffffu);          // global_addr[31:0]
        g0.w = (unsigned)((ga >> 32) & 0x1ffffffu) | (2u << 30); // addr[56:32], type=2
        i32x8 g1 = {0, 0, 0, 0, 0, 0, 0, 0};
        g1[0] = (1 << 16)        // data_size = 2 bytes
              | (1 << 20)        // pad_enable
              | (3 << 22)        // pad_interval: 16 dwords (64B) between pads
              | (1 << 25);       // pad_amount: 2 dwords (8B)
        g1[1] = (256 & 0xffff) << 16;                 // tensor_dim0 = 256 (lo16)
        g1[2] = (64 & 0xffff) << 16;                  // tensor_dim0 hi=0 | tensor_dim1=64
        g1[3] = (256 << 16);                          // tensor_dim1 hi=0 | tile_dim0=256
        g1[4] = 64;                                   // tile_dim1 = 64
        g1[5] = 256;                                  // tensor_dim0_stride = 256
        i32x4 z4 = {0, 0, 0, 0};
#if defined(__clang_major__) && __clang_major__ >= 23
        i32x8 z8 = {0, 0, 0, 0, 0, 0, 0, 0};
        __builtin_amdgcn_tensor_load_to_lds(g0, g1, z4, z4, z8, 0);
#else
        __builtin_amdgcn_tensor_load_to_lds(g0, g1, z4, z4, 0);
#endif
        __builtin_amdgcn_s_wait_tensorcnt(0);
    }
#else
    // cooperative fallback: same padded layout, dword copies
    for (int e = tid; e < 64 * 128; e += 128) {
        const int n  = e >> 7;
        const int kp = e & 127;                       // dword index within row data
        const unsigned w2 = ((const unsigned*)(gsrc + (size_t)n * 256))[kp];
        *(unsigned*)((char*)Wsh + n * 576 + (kp >> 4) * 72 + (kp & 15) * 4) = w2;
    }
#endif
}

// ---------------------------------------------------------------------------
// 3x3 conv (pad=1) as implicit GEMM over 9 taps. f16 WMMA, fp32 accumulate.
// Block = 128 thr = 4 waves; tile 64M x 64N; wave w -> rows [16w,16w+16),
// 4 N-subtiles -> 4 accumulators, 4 WMMA per K-step.
// wgt9: f16, layout [9][256 out][256 in] (tap-major, pre-converted).
// ---------------------------------------------------------------------------
template <int H>
__global__ __launch_bounds__(128)
void conv3x3_wmma_kernel(const float* __restrict__ src,
                         const _Float16* __restrict__ wgt9,
                         const float* __restrict__ bias,
                         float* __restrict__ dst, int relu)
{
    constexpr int W = H, HW = H * H, C = 256, Mtot = 2 * HW;
    const int m0   = blockIdx.x * 64;
    const int nBlk = blockIdx.y * 64;
    const int tid  = threadIdx.x;
    const int wave = tid >> 5;
    const int lane = tid & 31;
    const int half = lane >> 4;
    const int l16  = lane & 15;

    __shared__ _Float16 Ash[64 * LDSP];
    __shared__ _Float16 Wsh[64 * WROW];

    v8f acc[4];
#pragma unroll
    for (int nt = 0; nt < 4; ++nt) acc[nt] = (v8f){};

    for (int tap = 0; tap < 9; ++tap) {
        const int dy = tap / 3 - 1;
        const int dx = tap % 3 - 1;

        __syncthreads();   // all waves done reading previous tap's Wsh
        stage_w_tap(Wsh, wgt9 + (size_t)(tap * C + nBlk) * C, tid);
        __syncthreads();   // slab visible to all waves

        for (int kc = 0; kc < C; kc += 32) {
            // stage A (64 m x 32 k); consecutive threads -> consecutive m
            for (int e = tid; e < 64 * 32; e += 128) {
                const int m = e & 63;
                const int k = e >> 6;
                float v = 0.f;
                const int mg = m0 + m;
                if (mg < Mtot) {
                    const int b  = mg >= HW;
                    const int sp = mg & (HW - 1);
                    const int y  = sp / W;
                    const int x  = sp & (W - 1);
                    const int iy = y + dy, ix = x + dx;
                    if (iy >= 0 && iy < H && ix >= 0 && ix < W)
                        v = src[(size_t)(b * C + kc + k) * HW + iy * W + ix];
                }
                Ash[m * LDSP + k] = (_Float16)v;
            }
            __syncthreads();

            FragU a;
#pragma unroll
            for (int j = 0; j < 8; ++j)
                a.u[j] = *(const unsigned int*)&Ash[(wave * 16 + l16) * LDSP + a_koff(j, half)];
            const int wchunk = (kc >> 5) * 36;
#pragma unroll
            for (int nt = 0; nt < 4; ++nt) {
                FragU b;
#pragma unroll
                for (int j = 0; j < 8; ++j)
                    b.u[j] = *(const unsigned int*)&Wsh[(nt * 16 + l16) * WROW + wchunk + b_koff(j, half)];
                acc[nt] = __builtin_amdgcn_wmma_f32_16x16x32_f16(
                              false, a.v, false, b.v, (short)0, acc[nt], false, false);
            }
            __syncthreads();
        }
    }

#pragma unroll
    for (int nt = 0; nt < 4; ++nt) {
        const int n  = nBlk + nt * 16 + l16;
        const float bv = bias ? bias[n] : 0.f;
#pragma unroll
        for (int v = 0; v < 8; ++v) {
            const int mg = m0 + wave * 16 + v + half * 8;
            if (mg < Mtot) {
                const int b  = mg >= HW;
                const int sp = mg & (HW - 1);
                float o = acc[nt][v] + bv;
                if (relu) o = fmaxf(o, 0.f);
                dst[(size_t)(b * C + n) * HW + sp] = o;
            }
        }
    }
}

// ---------------------------------------------------------------------------
// Deformable 3x3 conv. Forward offsets reduce to py = y + oi[2k+1],
// px = x + oi[2k]. Bilinear params per (m, tap) computed once into LDS.
// ReLU fused at store, no bias.
// ---------------------------------------------------------------------------
template <int H>
__global__ __launch_bounds__(128)
void deform3x3_wmma_kernel(const float* __restrict__ src,
                           const float* __restrict__ offs,
                           const _Float16* __restrict__ wgt9,
                           float* __restrict__ dst)
{
    constexpr int W = H, HW = H * H, C = 256, Mtot = 2 * HW;
    const int m0   = blockIdx.x * 64;
    const int nBlk = blockIdx.y * 64;
    const int tid  = threadIdx.x;
    const int wave = tid >> 5;
    const int lane = tid & 31;
    const int half = lane >> 4;
    const int l16  = lane & 15;

    __shared__ _Float16 Ash[64 * LDSP];
    __shared__ _Float16 Wsh[64 * WROW];
    __shared__ int   Iy0[64], Ix0[64], Ib[64];
    __shared__ float Pwy[64], Pwx[64];

    v8f acc[4];
#pragma unroll
    for (int nt = 0; nt < 4; ++nt) acc[nt] = (v8f){};

    for (int tap = 0; tap < 9; ++tap) {
        __syncthreads();   // Wsh + param arrays free
        stage_w_tap(Wsh, wgt9 + (size_t)(tap * C + nBlk) * C, tid);

        // per-(m, tap) bilinear parameters
        for (int m = tid; m < 64; m += 128) {
            const int mg = m0 + m;
            int y0 = -2, x0 = -2, b = 0;
            float wy = 0.f, wx = 0.f;
            if (mg < Mtot) {
                b = mg >= HW;
                const int sp = mg & (HW - 1);
                const int y = sp / W;
                const int x = sp & (W - 1);
                const float py = (float)y + offs[(size_t)(b * 18 + 2 * tap + 1) * HW + sp];
                const float px = (float)x + offs[(size_t)(b * 18 + 2 * tap + 0) * HW + sp];
                const float y0f = floorf(py), x0f = floorf(px);
                wy = py - y0f; wx = px - x0f;
                y0 = (int)y0f; x0 = (int)x0f;
            }
            Iy0[m] = y0; Ix0[m] = x0; Pwy[m] = wy; Pwx[m] = wx; Ib[m] = b;
        }
        __syncthreads();

        for (int kc = 0; kc < C; kc += 32) {
            for (int e = tid; e < 64 * 32; e += 128) {
                const int m = e & 63;
                const int k = e >> 6;
                float v = 0.f;
                if (m0 + m < Mtot) {
                    const int y0 = Iy0[m], x0 = Ix0[m];
                    const float wy = Pwy[m], wx = Pwx[m];
                    const float* img = src + (size_t)(Ib[m] * C + kc + k) * HW;
                    auto g = [&](int yi, int xi) -> float {
                        if (yi < 0 || yi >= H || xi < 0 || xi >= W) return 0.f;
                        return img[yi * W + xi];
                    };
                    v = g(y0, x0)         * (1.f - wy) * (1.f - wx)
                      + g(y0, x0 + 1)     * (1.f - wy) * wx
                      + g(y0 + 1, x0)     * wy         * (1.f - wx)
                      + g(y0 + 1, x0 + 1) * wy         * wx;
                }
                Ash[m * LDSP + k] = (_Float16)v;
            }
            __syncthreads();

            FragU a;
#pragma unroll
            for (int j = 0; j < 8; ++j)
                a.u[j] = *(const unsigned int*)&Ash[(wave * 16 + l16) * LDSP + a_koff(j, half)];
            const int wchunk = (kc >> 5) * 36;
#pragma unroll
            for (int nt = 0; nt < 4; ++nt) {
                FragU b;
#pragma unroll
                for (int j = 0; j < 8; ++j)
                    b.u[j] = *(const unsigned int*)&Wsh[(nt * 16 + l16) * WROW + wchunk + b_koff(j, half)];
                acc[nt] = __builtin_amdgcn_wmma_f32_16x16x32_f16(
                              false, a.v, false, b.v, (short)0, acc[nt], false, false);
            }
            __syncthreads();
        }
    }

#pragma unroll
    for (int nt = 0; nt < 4; ++nt) {
        const int n = nBlk + nt * 16 + l16;
#pragma unroll
        for (int v = 0; v < 8; ++v) {
            const int mg = m0 + wave * 16 + v + half * 8;
            if (mg < Mtot) {
                const int b  = mg >= HW;
                const int sp = mg & (HW - 1);
                dst[(size_t)(b * C + n) * HW + sp] = fmaxf(acc[nt][v], 0.f);
            }
        }
    }
}

// ---------------------------------------------------------------------------
// 1x1 conv via WMMA. Block 128 = 4 waves; wave w -> M subtile w. N padded to
// 16-tiles (grid.y). Optional residual add (offs_refine = conv + oi).
// wgtF16: f16 [Nout][256] pre-converted.
// ---------------------------------------------------------------------------
template <int H>
__global__ __launch_bounds__(128)
void conv1x1_wmma_kernel(const float* __restrict__ src,
                         const _Float16* __restrict__ wgtF16,
                         const float* __restrict__ bias,
                         const float* __restrict__ add,
                         float* __restrict__ dst, int Nout)
{
    constexpr int HW = H * H, C = 256, Mtot = 2 * HW;
    const int m0   = blockIdx.x * 64;
    const int n0   = blockIdx.y * 16;
    const int tid  = threadIdx.x;
    const int wave = tid >> 5;
    const int lane = tid & 31;
    const int half = lane >> 4;
    const int l16  = lane & 15;

    __shared__ _Float16 Ash[64 * LDSP];
    __shared__ _Float16 Bsh[16 * LDSP];

    v8f acc = {};
    for (int kc = 0; kc < C; kc += 32) {
        for (int e = tid; e < 64 * 32; e += 128) {
            const int m = e & 63;
            const int k = e >> 6;
            float v = 0.f;
            const int mg = m0 + m;
            if (mg < Mtot) {
                const int b  = mg >= HW;
                const int sp = mg & (HW - 1);
                v = src[(size_t)(b * C + kc + k) * HW + sp];
            }
            Ash[m * LDSP + k] = (_Float16)v;
        }
        for (int e = tid; e < 256; e += 128) {
            const int n  = e >> 4;
            const int kp = e & 15;
            unsigned int w2 = 0;
            if (n0 + n < Nout)
                w2 = ((const unsigned int*)(wgtF16 + (size_t)(n0 + n) * C + kc))[kp];
            *(unsigned int*)&Bsh[n * LDSP + kp * 2] = w2;
        }
        __syncthreads();

        FragU a, b;
#pragma unroll
        for (int j = 0; j < 8; ++j) {
            a.u[j] = *(const unsigned int*)&Ash[(wave * 16 + l16) * LDSP + a_koff(j, half)];
            b.u[j] = *(const unsigned int*)&Bsh[l16 * LDSP + b_koff(j, half)];
        }
        acc = __builtin_amdgcn_wmma_f32_16x16x32_f16(
                  false, a.v, false, b.v, (short)0, acc, false, false);
        __syncthreads();
    }

    const int n = n0 + l16;
    if (n < Nout) {
        const float bv = bias ? bias[n] : 0.f;
#pragma unroll
        for (int v = 0; v < 8; ++v) {
            const int mg = m0 + wave * 16 + v + half * 8;
            if (mg < Mtot) {
                const int b  = mg >= HW;
                const int sp = mg & (HW - 1);
                float o = acc[v] + bv;
                if (add) o += add[(size_t)(b * Nout + n) * HW + sp];
                dst[(size_t)(b * Nout + n) * HW + sp] = o;
            }
        }
    }
}

// ---------------------------------------------------------------------------
// GroupNorm (32 groups, C=256 -> 8 ch/group), B=2, division-free.
// ---------------------------------------------------------------------------
__global__ __launch_bounds__(256)
void gn_stats_kernel(const float* __restrict__ src, float* __restrict__ stats, int HW)
{
    const int bg = blockIdx.x;           // b*32 + g
    const int b = bg >> 5, g = bg & 31;
    const float* base = src + (size_t)(b * 256 + g * 8) * HW;
    float s = 0.f, ss = 0.f;
    for (int c = 0; c < 8; ++c)
        for (int sp = threadIdx.x; sp < HW; sp += 256) {
            const float v = base[c * HW + sp];
            s += v; ss += v * v;
        }
    __shared__ float sh[256], sh2[256];
    sh[threadIdx.x] = s; sh2[threadIdx.x] = ss;
    __syncthreads();
    for (int o = 128; o > 0; o >>= 1) {
        if (threadIdx.x < o) {
            sh[threadIdx.x]  += sh[threadIdx.x + o];
            sh2[threadIdx.x] += sh2[threadIdx.x + o];
        }
        __syncthreads();
    }
    if (threadIdx.x == 0) {
        const float N = 8.f * (float)HW;
        const float mean = sh[0] / N;
        const float var  = sh2[0] / N - mean * mean;
        stats[bg * 2]     = mean;
        stats[bg * 2 + 1] = rsqrtf(var + 1e-5f);
    }
}

__global__ __launch_bounds__(256)
void gn_apply_kernel(const float* __restrict__ src, const float* __restrict__ stats,
                     const float* __restrict__ gamma, const float* __restrict__ beta,
                     float* __restrict__ dst, int HW)
{
    const int bc = blockIdx.y;           // b*256 + c
    const int b = bc >> 8, c = bc & 255, g = c >> 3;
    const int sp = blockIdx.x * 256 + threadIdx.x;
    if (sp < HW) {
        const float mean = stats[(b * 32 + g) * 2];
        const float rstd = stats[(b * 32 + g) * 2 + 1];
        const size_t i = (size_t)(b * 256 + c) * HW + sp;
        dst[i] = fmaxf((src[i] - mean) * rstd * gamma[c] + beta[c], 0.f);
    }
}

// ---------------------------------------------------------------------------
// Weight pre-conversion: OIHW fp32 -> tap-major f16 [9][Cout][Cin]; 1x1 copy.
// ---------------------------------------------------------------------------
__global__ __launch_bounds__(256)
void conv_w_to_f16_kernel(const float* __restrict__ w, _Float16* __restrict__ out)
{
    const int CC = 256 * 256;
    const size_t total = (size_t)CC * 9;
    for (size_t i = (size_t)blockIdx.x * 256 + threadIdx.x; i < total;
         i += (size_t)gridDim.x * 256) {
        const int tap = (int)(i / CC);
        const int r   = (int)(i - (size_t)tap * CC);   // o*256 + in
        out[i] = (_Float16)w[(size_t)r * 9 + tap];
    }
}

__global__ __launch_bounds__(256)
void w1x1_to_f16_kernel(const float* __restrict__ w, _Float16* __restrict__ out, int n)
{
    const int i = blockIdx.x * 256 + threadIdx.x;
    if (i < n) out[i] = (_Float16)w[i];
}

// ---------------------------------------------------------------------------
// Host orchestration
// ---------------------------------------------------------------------------
struct Ctx {
    const float* P[40];
    const _Float16* wc[9];   // cw0,cw1,cw2, rw0,rw1,rw2, iw1, dcw, drw (tap-major f16)
    const _Float16* w1[3];   // iw2, fw, lw (f16)
    float *T1, *T2, *CLS, *REG, *STATS;
    float* outF;
    hipStream_t stream;
};

template <int H>
static void run_level(const Ctx& c, const float* feat,
                      size_t lgOff, size_t oiOff, size_t orOff)
{
    constexpr int HW = H * H;
    constexpr int Mtot = 2 * HW;
    const int MtB = (Mtot + 63) / 64;
    const dim3 gConv(MtB, 4), bConv(128);
    const dim3 gGN((HW + 255) / 256, 512), bGN(256);
    hipStream_t s = c.stream;

    auto conv3 = [&](const float* src, const _Float16* w, const float* b, float* d, int relu) {
        conv3x3_wmma_kernel<H><<<gConv, bConv, 0, s>>>(src, w, b, d, relu);
    };
    auto gnrelu = [&](const float* src, const float* g, const float* be, float* d) {
        gn_stats_kernel<<<dim3(64), dim3(256), 0, s>>>(src, c.STATS, HW);
        gn_apply_kernel<<<gGN, bGN, 0, s>>>(src, c.STATS, g, be, d, HW);
    };
    auto c1x1 = [&](const float* src, const _Float16* w, const float* b,
                    const float* add, float* d, int Nout) {
        conv1x1_wmma_kernel<H><<<dim3(MtB, (Nout + 15) / 16), bConv, 0, s>>>(
            src, w, b, add, d, Nout);
    };
    auto branch = [&](int wbase, int pbase, float* outBuf) {
        const float* cur = feat;
        for (int i = 0; i < 3; ++i) {
            conv3(cur, c.wc[wbase + i], c.P[pbase + 4 * i + 1], c.T1, 0);
            float* dstBuf = (i == 2) ? outBuf : c.T2;
            gnrelu(c.T1, c.P[pbase + 4 * i + 2], c.P[pbase + 4 * i + 3], dstBuf);
            cur = dstBuf;
        }
    };

    branch(0, 5, c.CLS);    // cls tower
    branch(3, 17, c.REG);   // reg tower

    // offset-init: oi = iw2( relu(iw1 * reg_f) )
    conv3(c.REG, c.wc[6], c.P[32], c.T1, 1);
    c1x1(c.T1, c.w1[0], c.P[34], nullptr, c.outF + oiOff, 18);

    // cls logits: lw( relu(deform(cls_f, oi, dcw)) )
    deform3x3_wmma_kernel<H><<<gConv, bConv, 0, s>>>(c.CLS, c.outF + oiOff, c.wc[7], c.T1);
    c1x1(c.T1, c.w1[2], c.P[38], nullptr, c.outF + lgOff, 80);

    // offs_refine: fw( relu(deform(reg_f, oi, drw)) ) + oi
    deform3x3_wmma_kernel<H><<<gConv, bConv, 0, s>>>(c.REG, c.outF + oiOff, c.wc[8], c.T2);
    c1x1(c.T2, c.w1[1], c.P[36], c.outF + oiOff, c.outF + orOff, 18);
}

extern "C" void kernel_launch(void* const* d_in, const int* in_sizes, int n_in,
                              void* d_out, int out_size, void* d_ws, size_t ws_size,
                              hipStream_t stream)
{
    (void)in_sizes; (void)n_in; (void)out_size; (void)ws_size;

    Ctx c{};
    c.stream = stream;
    const float* feat[5];
    for (int i = 0; i < 5; ++i) feat[i] = (const float*)d_in[i];
    for (int i = 5; i < 39; ++i) c.P[i] = (const float*)d_in[i];

    static const int Hs[5] = {64, 32, 16, 8, 4};
    const size_t BUF = (size_t)2 * 256 * 64 * 64;   // floats, sized for level 0
    float* base = (float*)d_ws;
    c.T1 = base; c.T2 = c.T1 + BUF; c.CLS = c.T2 + BUF; c.REG = c.CLS + BUF;
    c.STATS = c.REG + BUF;                           // 128 floats

    // f16 weight region (dword-aligned: follows float data)
    _Float16* wh = (_Float16*)(c.STATS + 128);
    const size_t W3SZ = (size_t)9 * 256 * 256;
    static const int wcSrc[9] = {5, 9, 13, 17, 21, 25, 31, 29, 30};
    for (int i = 0; i < 9; ++i) { c.wc[i] = wh; wh += W3SZ; }
    static const int w1Src[3] = {33, 35, 37};
    static const int w1N[3]   = {18 * 256, 18 * 256, 80 * 256};
    for (int i = 0; i < 3; ++i) { c.w1[i] = wh; wh += w1N[i]; }

    // pre-convert weights (stream-ordered, once per call; graph-capture safe)
    for (int i = 0; i < 9; ++i)
        conv_w_to_f16_kernel<<<dim3(2304), dim3(256), 0, stream>>>(
            c.P[wcSrc[i]], (_Float16*)c.wc[i]);
    for (int i = 0; i < 3; ++i)
        w1x1_to_f16_kernel<<<dim3((w1N[i] + 255) / 256), dim3(256), 0, stream>>>(
            c.P[w1Src[i]], (_Float16*)c.w1[i], w1N[i]);

    c.outF = (float*)d_out;
    size_t lgOff[5], oiOff[5], orOff[5], off = 0;
    for (int i = 0; i < 5; ++i) { lgOff[i] = off; off += (size_t)2 * 80 * Hs[i] * Hs[i]; }
    for (int i = 0; i < 5; ++i) { oiOff[i] = off; off += (size_t)2 * 18 * Hs[i] * Hs[i]; }
    for (int i = 0; i < 5; ++i) { orOff[i] = off; off += (size_t)2 * 18 * Hs[i] * Hs[i]; }

    for (int lv = 0; lv < 5; ++lv) {
        switch (Hs[lv]) {
        case 64: run_level<64>(c, feat[lv], lgOff[lv], oiOff[lv], orOff[lv]); break;
        case 32: run_level<32>(c, feat[lv], lgOff[lv], oiOff[lv], orOff[lv]); break;
        case 16: run_level<16>(c, feat[lv], lgOff[lv], oiOff[lv], orOff[lv]); break;
        case 8:  run_level<8>(c, feat[lv], lgOff[lv], oiOff[lv], orOff[lv]); break;
        case 4:  run_level<4>(c, feat[lv], lgOff[lv], oiOff[lv], orOff[lv]); break;
        }
    }
}
